// BoostedNeuralLDPCDecoder_31714038513711
// MI455X (gfx1250) — matrix operand order repaired and checked by reading, not compiled, not run on hardware
//
#include <hip/hip_runtime.h>
#include <hip/hip_bf16.h>
#include <stdint.h>

// ---------------------------------------------------------------------------
// Quantized min-sum LDPC decoder for MI455X (gfx1250), single kernel.
// One workgroup (768 threads = 24 waves) per batch element b; all iterations
// run inside the kernel with per-b state resident in LDS (~316KB of the
// CDNA5 WGP's 320KB).
//   c2v messages : int8 (value*2; quantized halves in [-7.5,7.5]) -> exact
//   tot einsum   : v_wmma_f32_16x16x32_bf16, B = 0.5-hot matrix -> EXACT
//                  ((2v) * 0.5 = v, all multiples of 0.5, f32 accum)
//   B fragments  : iteration-invariant -> prebuilt ONCE (scaled by 0.5),
//                  cached in LDS, re-loaded as 2x ds_load_b128 per tile
//   A fragments  : built once per K-step (bfe+cvt+pack only), reused 5x
//   edge tables  : (shift | vn<<16) packed -> 1 ds_load_b32 per edge
// ---------------------------------------------------------------------------

typedef __attribute__((ext_vector_type(16))) __bf16       v16bf;
typedef __attribute__((ext_vector_type(8)))  float        v8f;
typedef __attribute__((ext_vector_type(4)))  unsigned int v4u;

union BFrag { v16bf v; v4u q[2]; };

#define B_    32
#define Z_    384
#define N_    68
#define M_    46
#define DC_   7
#define E_    322
#define EP_   360          // padded c2v row stride (bytes); 8-aligned, bank-friendly
#define KPAD_ 352          // K padded to multiple of 32 for WMMA (11 * 32)
#define NKB_  11
#define NNT_  5            // ceil(N/16) (N-tiles per strip)
#define NFRAG_ (NKB_ * NNT_)   // 55 cached B fragments
#define SP_   83           // S row stride in floats (odd mod 64 -> conflict-free)
#define NZT_  24           // Z/16  (z-strips == waves)
#define NTHREADS_ 768      // 24 waves: 1 z-strip each; 17664 % 768 == 0
#define NWAVES_ 24

// LDS layout (dynamic shared):
#define C2V_OFF   0
#define C2V_BYTES (Z_ * EP_)                 // 138240
#define S_OFF     (C2V_OFF + C2V_BYTES)      // 138240
#define S_BYTES   (Z_ * SP_ * 4)             // 127488
#define BF_OFF    (S_OFF + S_BYTES)          // 265728 (128-aligned)
#define BF_BYTES  (NFRAG_ * 32 * 32)         // 56320 (32B per lane per frag)
#define VN_OFF    (BF_OFF + BF_BYTES)        // 322048 (8-aligned)
#define ET_OFF    (VN_OFF + KPAD_)           // 322400 (4-aligned)
#define LDS_BYTES 323712                     // <= 327680 (320KB per WGP)

__global__ __launch_bounds__(NTHREADS_)
void ldpc_minsum_kernel(const float* __restrict__ xa_g,
                        const float* __restrict__ cw_g,
                        const int*   __restrict__ vn_g,
                        const int*   __restrict__ sh_g,
                        float*       __restrict__ out_g,
                        int iters)
{
    extern __shared__ char lds[];
    int8_t*   c2v   = (int8_t*)  (lds + C2V_OFF);   // [Z_][EP_], value = 2*message
    float*    S     = (float*)   (lds + S_OFF);     // [Z_][SP_], belief xa+tot
    v4u*      bfrag = (v4u*)     (lds + BF_OFF);    // 55 cached 0.5-hot B fragments
    uint8_t*  vn8   = (uint8_t*) (lds + VN_OFF);    // [KPAD_], pad 0xFF (init only)
    uint32_t* et32  = (uint32_t*)(lds + ET_OFF);    // [E_]: shift | (vn<<16)

    const int tid  = threadIdx.x;
    const int b    = blockIdx.x;
    const int lane = tid & 31;
    const int wave = tid >> 5;           // 0..23 == z-strip id
    const int ml   = lane & 15;
    const int hi   = (lane >> 4) & 1;

    const float* xab = xa_g + (size_t)b * (Z_ * N_);

    // ---- init: zero c2v (incl. K padding), load edge tables ----
    for (int i = tid; i < C2V_BYTES / 4; i += NTHREADS_)
        ((int*)c2v)[i] = 0;
    if (tid < KPAD_)
        vn8[tid] = (tid < E_) ? (uint8_t)vn_g[tid] : (uint8_t)0xFF;
    if (tid < E_)
        et32[tid] = (uint32_t)sh_g[tid] | ((uint32_t)vn_g[tid] << 16);
    // warm L2/WGP$ for the xa slab (global_prefetch_b8)
    __builtin_prefetch(xab + tid * 8, 0, 1);
    __syncthreads();

    // ---- prebuild the 55 iteration-invariant 0.5-hot B fragments in LDS ----
    // B k order used here (contiguous, same as A): element j <-> K = K0 + j + 16*hi; column n = ml
    // scale 0.5 folded into B so A needs no scaling: (2v) * 0.5 = v (exact)
    for (int f = wave; f < NFRAG_; f += NWAVES_) {
        const int kb = f / NNT_;
        const int nb = f % NNT_;
        const int K0 = kb * 32;
        const int n  = nb * 16 + ml;
        uint64_t q0 = *(const uint64_t*)(vn8 + K0 + hi * 16);
        uint64_t q1 = *(const uint64_t*)(vn8 + K0 + hi * 16 + 8);
        BFrag bu;
        #pragma unroll
        for (int j = 0; j < 8; ++j) {
            bu.v[j]     = ((int)((q0 >> (8 * j)) & 0xFF) == n) ? (__bf16)0.5f : (__bf16)0.0f;
            bu.v[j + 8] = ((int)((q1 >> (8 * j)) & 0xFF) == n) ? (__bf16)0.5f : (__bf16)0.0f;
        }
        v4u* dst = bfrag + ((size_t)f * 32 + lane) * 2;
        dst[0] = bu.q[0];
        dst[1] = bu.q[1];
    }
    __syncthreads();

    for (int it = 0; it <= iters; ++it) {
        const bool finalPass = (it == iters);

        // ============ phase A: tot via WMMA; S = xa + tot (or write out) ============
        {
            const int z0 = wave * 16;                      // this wave's z-strip
            const int8_t* crow = c2v + (z0 + ml) * EP_;

            v8f acc[NNT_];
            #pragma unroll
            for (int nb = 0; nb < NNT_; ++nb)
                acc[nb] = (v8f){0.f, 0.f, 0.f, 0.f, 0.f, 0.f, 0.f, 0.f};

            for (int kb = 0; kb < NKB_; ++kb) {
                const int K0 = kb * 32;

                // fetch the 5 cached B fragments (2x ds_load_b128 each)
                BFrag bu[NNT_];
                #pragma unroll
                for (int nb = 0; nb < NNT_; ++nb) {
                    const v4u* src = bfrag + ((size_t)(kb * NNT_ + nb) * 32 + lane) * 2;
                    bu[nb].q[0] = src[0];
                    bu[nb].q[1] = src[1];
                }

                // A fragment (built ONCE per K-step, reused for all 5 N-tiles)
                // A k order used here (contiguous, same as B): k = j + 16*hi; no scaling needed
                uint64_t r0 = *(const uint64_t*)(crow + K0 + hi * 16);
                uint64_t r1 = *(const uint64_t*)(crow + K0 + hi * 16 + 8);
                v16bf a;
                #pragma unroll
                for (int j = 0; j < 8; ++j) {
                    a[j]     = (__bf16)(float)(int8_t)(r0 >> (8 * j));
                    a[j + 8] = (__bf16)(float)(int8_t)(r1 >> (8 * j));
                }

                // 5 back-to-back WMMAs (distinct D, shared read-only A/B -> no hazard)
                #pragma unroll
                for (int nb = 0; nb < NNT_; ++nb)
                    acc[nb] = __builtin_amdgcn_wmma_f32_16x16x32_bf16(
                                  false, a, false, bu[nb].v, (short)0, acc[nb], false, false);
            }

            // C/D layout: VGPR r, lanes 0-15 -> M=r, lanes 16-31 -> M=r+8
            if (!finalPass) {
                #pragma unroll
                for (int nb = 0; nb < NNT_; ++nb) {
                    const int n  = nb * 16 + ml;
                    const int nc = (n < N_) ? n : 0;       // clamp: loads always in-bounds
                    #pragma unroll
                    for (int r = 0; r < 8; ++r) {
                        const int z = z0 + r + hi * 8;
                        float xv = xab[z * N_ + nc];       // unconditional load
                        xv = (n < N_) ? xv : 0.0f;         // branchless select
                        S[z * SP_ + n] = xv + acc[nb][r];
                    }
                }
            } else {
                #pragma unroll
                for (int nb = 0; nb < NNT_; ++nb) {
                    const int n = nb * 16 + ml;
                    if (n < N_) {
                        float* ob = out_g + (size_t)b * (N_ * Z_) + (size_t)n * Z_;
                        #pragma unroll
                        for (int r = 0; r < 8; ++r) {
                            const int z = z0 + r + hi * 8;
                            ob[z] = xab[z * N_ + n] + acc[nb][r];
                        }
                    }
                }
            }
        }
        if (finalPass) break;          // uniform across the workgroup
        __syncthreads();

        // ================= phase B: check-node min-sum update =================
        const float w = cw_g[it];      // uniform scalar load
        for (int t = tid; t < M_ * Z_; t += NTHREADS_) {
            const int zp = t % Z_;     // lifted-domain row
            const int m  = t / Z_;
            const int e0 = m * DC_;

            float aabs[DC_]; int neg[DC_]; int zl[DC_];
            #pragma unroll
            for (int e = 0; e < DC_; ++e) {
                const uint32_t et = et32[e0 + e];          // shift | (vn<<16)
                int z = zp + (int)(et & 0xFFFF);
                if (z >= Z_) z -= Z_;
                const int vn = (int)(et >> 16);
                float v = S[z * SP_ + vn] - 0.5f * (float)c2v[z * EP_ + e0 + e];
                v = fminf(fmaxf(v, -20.0f), 20.0f);     // LLR clip
                neg[e]  = (v < 0.0f) ? 1 : 0;
                aabs[e] = fabsf(v);
                zl[e]   = z;
            }
            // min1 + first index (strict < keeps first occurrence, like segment_min)
            float min1 = aabs[0]; int first = 0;
            #pragma unroll
            for (int e = 1; e < DC_; ++e)
                if (aabs[e] < min1) { min1 = aabs[e]; first = e; }
            float min2 = __builtin_inff();
            #pragma unroll
            for (int e = 0; e < DC_; ++e)
                if (e != first) min2 = fminf(min2, aabs[e]);
            int negtot = 0;
            #pragma unroll
            for (int e = 0; e < DC_; ++e) negtot += neg[e];

            #pragma unroll
            for (int e = 0; e < DC_; ++e) {
                const float ext = (e == first) ? min2 : min1;
                const float sg  = ((negtot - neg[e]) & 1) ? -1.0f : 1.0f;
                const float x   = (w * sg) * ext;
                float r = rintf(x * 2.0f);               // round half-to-even
                r = fminf(fmaxf(r, -15.0f), 15.0f);      // clip to +-7.5 (x2)
                c2v[zl[e] * EP_ + e0 + e] = (int8_t)r;   // in-place update (unique owner)
            }
        }
        __syncthreads();
    }
}

extern "C" void kernel_launch(void* const* d_in, const int* in_sizes, int n_in,
                              void* d_out, int out_size, void* d_ws, size_t ws_size,
                              hipStream_t stream) {
    (void)n_in; (void)out_size; (void)d_ws; (void)ws_size;
    const float* xa = (const float*)d_in[0];
    const float* cw = (const float*)d_in[1];
    const int*   vn = (const int*)  d_in[2];
    // d_in[3] = cn_idx (repeat(arange(M), dc)) -- structure hardcoded
    const int*   sh = (const int*)  d_in[4];
    float* out = (float*)d_out;
    const int iters = in_sizes[1];   // cn_weights.shape[0]

    // allow > 64KB dynamic LDS (idempotent; not a stream op, capture-safe)
    (void)hipFuncSetAttribute(reinterpret_cast<const void*>(ldpc_minsum_kernel),
                              hipFuncAttributeMaxDynamicSharedMemorySize, LDS_BYTES);

    ldpc_minsum_kernel<<<dim3(B_), dim3(NTHREADS_), LDS_BYTES, stream>>>(
        xa, cw, vn, sh, out, iters);
}
